// ManoLayer_68839735821135
// MI455X (gfx1250) — matrix operand order repaired and checked by reading, not compiled, not run on hardware
//
#include <hip/hip_runtime.h>
#include <hip/hip_bf16.h>
#include <math.h>

// ---------------------------------------------------------------------------
// MANO forward for MI455X (gfx1250, wave32).
// Heavy GEMM (4096x145 @ 145x2334) runs on v_wmma_f32_16x16x32_f16 with fp32
// accumulate; skinning fused in the same block via LDS staging.
// ---------------------------------------------------------------------------

#define N_VERTS   778
#define N_JOINTS  16
#define KDIM      145          // 10 betas + 135 pose_map
#define KPAD      160          // 5 x 32
#define KSTEPS    5
#define NCOLS     (N_VERTS*3)  // 2334
#define VTILES    49           // ceil(2334/48)
#define NTILES    (VTILES*3)   // 147 column tiles of 16
#define BFRAG_ELEMS ((size_t)NTILES * KSTEPS * 32 * 16)

typedef _Float16 v16h __attribute__((ext_vector_type(16)));
typedef _Float16 h8v  __attribute__((ext_vector_type(8)));
typedef float    v8f  __attribute__((ext_vector_type(8)));

__constant__ int c_reorder[21] = {0,13,14,15,16,1,2,3,17,4,5,6,18,10,11,12,19,7,8,9,20};
__constant__ int c_tips[5]    = {745,317,444,556,673};

// ---------------------------------------------------------------------------
// Kernel 1: per-batch pose PCA (45x45), Rodrigues -> rot mats, and the
// WMMA A-matrix row (f16): [betas(10) | pose_map(135) | zeros(15)]
// ---------------------------------------------------------------------------
__global__ void k_prep_pose(const float* __restrict__ coeffs,
                            const float* __restrict__ betas,
                            const float* __restrict__ SC,        // 45x45
                            const float* __restrict__ hmean,     // 45
                            float* __restrict__ rotws,           // b x 16 x 9
                            _Float16* __restrict__ Xf16,         // b x 160
                            int b)
{
    int i = blockIdx.x * blockDim.x + threadIdx.x;
    if (i >= b) return;

    float pc[48];
#pragma unroll
    for (int t = 0; t < 48; ++t) pc[t] = coeffs[i * 48 + t];

    float fullpose[48];
    fullpose[0] = pc[0]; fullpose[1] = pc[1]; fullpose[2] = pc[2];
    for (int n = 0; n < 45; ++n) {
        float s = hmean[n];
#pragma unroll 5
        for (int m = 0; m < 45; ++m) s += pc[3 + m] * SC[m * 45 + n];
        fullpose[3 + n] = s;
    }

    _Float16* xr = Xf16 + (size_t)i * KPAD;
#pragma unroll
    for (int k = 0; k < 10; ++k) xr[k] = (_Float16)betas[i * 10 + k];
#pragma unroll
    for (int k = KDIM; k < KPAD; ++k) xr[k] = (_Float16)0.0f;

    for (int jnt = 0; jnt < 16; ++jnt) {
        float ax = fullpose[jnt * 3 + 0];
        float ay = fullpose[jnt * 3 + 1];
        float az = fullpose[jnt * 3 + 2];
        float ex = ax + 1e-8f, ey = ay + 1e-8f, ez = az + 1e-8f;
        float norm = sqrtf(ex * ex + ey * ey + ez * ez);
        float inv  = 1.0f / norm;
        float half = 0.5f * norm;
        float sh = sinf(half);
        float qw = cosf(half);
        float qx = sh * ax * inv, qy = sh * ay * inv, qz = sh * az * inv;
        float qn = rsqrtf(qw * qw + qx * qx + qy * qy + qz * qz);
        qw *= qn; qx *= qn; qy *= qn; qz *= qn;

        float w2 = qw * qw, x2 = qx * qx, y2 = qy * qy, z2 = qz * qz;
        float wx = qw * qx, wy = qw * qy, wz = qw * qz;
        float xy = qx * qy, xz = qx * qz, yz = qy * qz;
        float R[9];
        R[0] = w2 + x2 - y2 - z2; R[1] = 2.f*xy - 2.f*wz;     R[2] = 2.f*wy + 2.f*xz;
        R[3] = 2.f*wz + 2.f*xy;   R[4] = w2 - x2 + y2 - z2;   R[5] = 2.f*yz - 2.f*wx;
        R[6] = 2.f*xz - 2.f*wy;   R[7] = 2.f*wx + 2.f*yz;     R[8] = w2 - x2 - y2 + z2;

        float* rw = rotws + (size_t)i * 144 + jnt * 9;
#pragma unroll
        for (int q = 0; q < 9; ++q) rw[q] = R[q];

        if (jnt >= 1) {
#pragma unroll
            for (int r = 0; r < 3; ++r)
#pragma unroll
                for (int c = 0; c < 3; ++c)
                    xr[10 + (jnt - 1) * 9 + r * 3 + c] =
                        (_Float16)(R[r * 3 + c] - ((r == c) ? 1.0f : 0.0f));
        }
    }
}

// ---------------------------------------------------------------------------
// Kernel 2 (once): Jshape[16][3][10] = J_regressor @ shapedirs,
//                  Jtempl[16][3]    = J_regressor @ v_template
// ---------------------------------------------------------------------------
__global__ void k_jreduce(const float* __restrict__ JR,
                          const float* __restrict__ shapedirs,
                          const float* __restrict__ vtempl,
                          float* __restrict__ Jshape,   // 480
                          float* __restrict__ Jtempl)   // 48
{
    int t = threadIdx.x;
    if (t < 480) {
        int jnt = t / 30, rem = t % 30, d = rem / 10, k = rem % 10;
        float acc = 0.f;
        for (int v = 0; v < N_VERTS; ++v)
            acc += JR[jnt * N_VERTS + v] * shapedirs[(size_t)(v * 3 + d) * 10 + k];
        Jshape[t] = acc;
    } else if (t < 528) {
        int u = t - 480, jnt = u / 3, d = u % 3;
        float acc = 0.f;
        for (int v = 0; v < N_VERTS; ++v)
            acc += JR[jnt * N_VERTS + v] * vtempl[v * 3 + d];
        Jtempl[u] = acc;
    }
}

// ---------------------------------------------------------------------------
// Kernel 3 (once): pack B = [shapedirs | posedirs]^T (145 x 2334) into
// WMMA B-fragment order (f16), zero-padded to K=160, cols padded to 2352.
// B layout (32x16 f16): lanes 0-15 hold K 0..7,16..23, lanes 16-31 hold K 8..15,24..31.
// ---------------------------------------------------------------------------
__global__ void k_prepB(const float* __restrict__ shapedirs,
                        const float* __restrict__ posedirs,
                        _Float16* __restrict__ Bfrag)
{
    size_t t = (size_t)blockIdx.x * blockDim.x + threadIdx.x;
    if (t >= BFRAG_ELEMS) return;
    int h    = (int)(t & 15);
    int lane = (int)((t >> 4) & 31);
    int ks   = (int)((t >> 9) % KSTEPS);
    int nt   = (int)(t / (16 * 32 * KSTEPS));
    int c = nt * 16 + (lane & 15);
    int k = ks * 32 + ((h >> 3) << 4) + ((lane >> 4) << 3) + (h & 7);
    float v = 0.f;
    if (c < NCOLS && k < KDIM)
        v = (k < 10) ? shapedirs[(size_t)c * 10 + k]
                     : posedirs[(size_t)c * 135 + (k - 10)];
    Bfrag[t] = (_Float16)v;
}

// ---------------------------------------------------------------------------
// Kernel 4: per-batch joint regression + FK chain -> skinning matrices G
// (b x 16 x 12) and joint translations jtr (b x 16 x 3).
// Parent of joint i (i>=1) is 0 if i%3==1 else i-1.
// ---------------------------------------------------------------------------
__global__ void k_fk(const float* __restrict__ betas,
                     const float* __restrict__ rotws,
                     const float* __restrict__ Jshape,
                     const float* __restrict__ Jtempl,
                     float* __restrict__ Gws,     // b x 192
                     float* __restrict__ jtrws,   // b x 48
                     int b)
{
    int i = blockIdx.x * blockDim.x + threadIdx.x;
    if (i >= b) return;

    float bt[10];
#pragma unroll
    for (int k = 0; k < 10; ++k) bt[k] = betas[i * 10 + k];

    float jj[48];
#pragma unroll 4
    for (int q = 0; q < 48; ++q) {
        float s = Jtempl[q];
#pragma unroll
        for (int k = 0; k < 10; ++k) s += Jshape[q * 10 + k] * bt[k];
        jj[q] = s;
    }

    float R0[9], t0[3], Rp[9], tp[3];
    const float* rwb = rotws + (size_t)i * 144;
    float* gb  = Gws + (size_t)i * 192;
    float* jb  = jtrws + (size_t)i * 48;

    for (int jnt = 0; jnt < 16; ++jnt) {
        float Rl[9];
#pragma unroll
        for (int q = 0; q < 9; ++q) Rl[q] = rwb[jnt * 9 + q];

        float Rg[9], tg[3];
        if (jnt == 0) {
#pragma unroll
            for (int q = 0; q < 9; ++q) Rg[q] = Rl[q];
            tg[0] = jj[0]; tg[1] = jj[1]; tg[2] = jj[2];
        } else {
            int p = (jnt % 3 == 1) ? 0 : (jnt - 1);
            const float* PR = (p == 0) ? R0 : Rp;
            const float* PT = (p == 0) ? t0 : tp;
            float tl[3];
#pragma unroll
            for (int d = 0; d < 3; ++d) tl[d] = jj[jnt * 3 + d] - jj[p * 3 + d];
#pragma unroll
            for (int r = 0; r < 3; ++r) {
#pragma unroll
                for (int c = 0; c < 3; ++c)
                    Rg[r * 3 + c] = PR[r * 3 + 0] * Rl[0 * 3 + c] +
                                    PR[r * 3 + 1] * Rl[1 * 3 + c] +
                                    PR[r * 3 + 2] * Rl[2 * 3 + c];
                tg[r] = PR[r * 3 + 0] * tl[0] + PR[r * 3 + 1] * tl[1] +
                        PR[r * 3 + 2] * tl[2] + PT[r];
            }
        }

#pragma unroll
        for (int d = 0; d < 3; ++d) jb[jnt * 3 + d] = tg[d];

#pragma unroll
        for (int r = 0; r < 3; ++r) {
            float cg = Rg[r * 3 + 0] * jj[jnt * 3 + 0] +
                       Rg[r * 3 + 1] * jj[jnt * 3 + 1] +
                       Rg[r * 3 + 2] * jj[jnt * 3 + 2];
#pragma unroll
            for (int c = 0; c < 3; ++c) gb[jnt * 12 + r * 4 + c] = Rg[r * 3 + c];
            gb[jnt * 12 + r * 4 + 3] = tg[r] - cg;
        }

        if (jnt == 0) {
#pragma unroll
            for (int q = 0; q < 9; ++q) R0[q] = Rg[q];
            t0[0] = tg[0]; t0[1] = tg[1]; t0[2] = tg[2];
        }
#pragma unroll
        for (int q = 0; q < 9; ++q) Rp[q] = Rg[q];
        tp[0] = tg[0]; tp[1] = tg[1]; tp[2] = tg[2];
    }
}

// ---------------------------------------------------------------------------
// Kernel 5: fused WMMA GEMM (v_posed tile, 16 batches x 16 verts x 3 comps)
// + LDS-staged linear-blend skinning + output writeback.
// One wave (32 threads) per block; 15 v_wmma_f32_16x16x32_f16 per block.
// ---------------------------------------------------------------------------
__global__ void __launch_bounds__(32) k_gemm_skin(
    const _Float16* __restrict__ Xf16,   // b x 160
    const _Float16* __restrict__ Bfrag,  // fragment-ordered
    const float* __restrict__ vtempl,    // 2334
    const float* __restrict__ weights,   // 778 x 16
    const float* __restrict__ Gws,       // b x 192
    const float* __restrict__ trans,     // b x 3
    float* __restrict__ out,             // verts region: b x 778 x 3
    int b)
{
    const int lane = threadIdx.x;
    const int bt = blockIdx.x;           // batch tile (b/16)
    const int vt = blockIdx.y;           // vertex tile (49)
    const int b0 = bt * 16;
    const int v0 = vt * 16;

    __shared__ float vp[16][49];         // v_posed tile [batch-row][48 cols]+pad
    __shared__ float Wt[16][17];         // weights tile [vert][joint]+pad
    __shared__ float Gt[16][192];        // skinning mats [batch-row][16*12]

    v8f acc0 = {}, acc1 = {}, acc2 = {};

    int arow = b0 + (lane & 15);
    if (arow >= b) arow = b - 1;
    const _Float16* Xr = Xf16 + (size_t)arow * KPAD;
    const int koff = (lane >> 4) << 3;   // A: lanes 16-31 shift K by 8

#pragma unroll
    for (int ks = 0; ks < KSTEPS; ++ks) {
        h8v lo = *(const h8v*)(Xr + ks * 32 + koff);
        h8v hi = *(const h8v*)(Xr + ks * 32 + 16 + koff);
        v16h a = __builtin_shufflevector(lo, hi, 0, 1, 2, 3, 4, 5, 6, 7,
                                                 8, 9, 10, 11, 12, 13, 14, 15);
        const _Float16* Bp =
            Bfrag + ((((size_t)(vt * 3) * KSTEPS + ks) * 32 + lane) << 4);
        v16h bf0 = *(const v16h*)(Bp);
        v16h bf1 = *(const v16h*)(Bp + (size_t)KSTEPS * 32 * 16);
        v16h bf2 = *(const v16h*)(Bp + (size_t)2 * KSTEPS * 32 * 16);
        acc0 = __builtin_amdgcn_wmma_f32_16x16x32_f16(false, a, false, bf0,
                                                      (short)0, acc0, false, false);
        acc1 = __builtin_amdgcn_wmma_f32_16x16x32_f16(false, a, false, bf1,
                                                      (short)0, acc1, false, false);
        acc2 = __builtin_amdgcn_wmma_f32_16x16x32_f16(false, a, false, bf2,
                                                      (short)0, acc2, false, false);
    }

    // Scatter C fragments (16x16 f32: lanes 0-15 rows i, lanes 16-31 rows 8+i)
    const int N = lane & 15;
    const int mbase = (lane < 16) ? 0 : 8;
#pragma unroll
    for (int nt = 0; nt < 3; ++nt) {
        v8f acc = (nt == 0) ? acc0 : (nt == 1) ? acc1 : acc2;
        int l = nt * 16 + N;
        int c = vt * 48 + l;
        float tv = (c < NCOLS) ? vtempl[c] : 0.f;
#pragma unroll
        for (int i = 0; i < 8; ++i) vp[mbase + i][l] = acc[i] + tv;
    }

    // Stage weights and skinning matrices
    for (int idx = lane; idx < 256; idx += 32) {
        int vl = idx >> 4, j = idx & 15;
        int vg = v0 + vl;
        Wt[vl][j] = (vg < N_VERTS) ? weights[vg * 16 + j] : 0.f;
    }
    for (int idx = lane; idx < 3072; idx += 32) {
        int bl = idx / 192, r = idx % 192;
        int bg = b0 + bl; if (bg >= b) bg = b - 1;
        Gt[bl][r] = Gws[(size_t)bg * 192 + r];
    }
    __syncthreads();

    // Skinning: 256 (batch,vert) pairs over 32 lanes
    for (int p = lane; p < 256; p += 32) {
        int bl = p >> 4, vl = p & 15;
        int vg = v0 + vl, bg = b0 + bl;
        if (vg >= N_VERTS || bg >= b) continue;
        float x = vp[bl][vl * 3 + 0];
        float y = vp[bl][vl * 3 + 1];
        float z = vp[bl][vl * 3 + 2];
        float m[12];
#pragma unroll
        for (int q = 0; q < 12; ++q) m[q] = 0.f;
#pragma unroll
        for (int j = 0; j < 16; ++j) {
            float w = Wt[vl][j];
            const float* g = &Gt[bl][j * 12];
#pragma unroll
            for (int q = 0; q < 12; ++q) m[q] += w * g[q];
        }
        float ox = m[0] * x + m[1] * y + m[2]  * z + m[3];
        float oy = m[4] * x + m[5] * y + m[6]  * z + m[7];
        float oz = m[8] * x + m[9] * y + m[10] * z + m[11];
        size_t o = ((size_t)bg * N_VERTS + vg) * 3;
        out[o + 0] = (ox + trans[bg * 3 + 0]) * 1000.f;
        out[o + 1] = (oy + trans[bg * 3 + 1]) * 1000.f;
        out[o + 2] = (oz + trans[bg * 3 + 2]) * 1000.f;
    }
}

// ---------------------------------------------------------------------------
// Kernel 6: jtr = reorder(concat(joint translations, tip verts)) -> output tail
// ---------------------------------------------------------------------------
__global__ void k_jtr(const float* __restrict__ jtrws,
                      const float* __restrict__ trans,
                      float* __restrict__ out,
                      int b)
{
    int t = blockIdx.x * blockDim.x + threadIdx.x;
    if (t >= b * 21) return;
    int i = t / 21, o = t % 21;
    int src = c_reorder[o];
    size_t vbase = (size_t)b * N_VERTS * 3;
    float vals[3];
    if (src < 16) {
#pragma unroll
        for (int d = 0; d < 3; ++d)
            vals[d] = (jtrws[(size_t)i * 48 + src * 3 + d] + trans[i * 3 + d]) * 1000.f;
    } else {
        int vid = c_tips[src - 16];
#pragma unroll
        for (int d = 0; d < 3; ++d)
            vals[d] = out[((size_t)i * N_VERTS + vid) * 3 + d];  // already transformed
    }
#pragma unroll
    for (int d = 0; d < 3; ++d)
        out[vbase + ((size_t)i * 21 + o) * 3 + d] = vals[d];
}

// ---------------------------------------------------------------------------
extern "C" void kernel_launch(void* const* d_in, const int* in_sizes, int n_in,
                              void* d_out, int out_size, void* d_ws, size_t ws_size,
                              hipStream_t stream)
{
    const float* coeffs    = (const float*)d_in[0];   // b x 48
    const float* betas     = (const float*)d_in[1];   // b x 10
    const float* trans     = (const float*)d_in[2];   // b x 3
    const float* SC        = (const float*)d_in[3];   // 45 x 45
    const float* hmean     = (const float*)d_in[4];   // 45
    const float* shapedirs = (const float*)d_in[5];   // 778 x 3 x 10
    const float* posedirs  = (const float*)d_in[6];   // 778 x 3 x 135
    const float* vtempl    = (const float*)d_in[7];   // 778 x 3
    const float* JR        = (const float*)d_in[8];   // 16 x 778
    const float* weights   = (const float*)d_in[9];   // 778 x 16

    const int b = in_sizes[0] / 48;
    if (b <= 0) return;

    // Workspace carve-up (256B aligned regions)
    char* p = (char*)d_ws;
    auto take = [&](size_t bytes) {
        char* r = p;
        p += (bytes + 255) & ~(size_t)255;
        return r;
    };
    float*    Gws    = (float*)take((size_t)b * 192 * sizeof(float));
    float*    jtrws  = (float*)take((size_t)b * 48 * sizeof(float));
    float*    rotws  = (float*)take((size_t)b * 144 * sizeof(float));
    float*    Jshape = (float*)take(480 * sizeof(float));
    float*    Jtempl = (float*)take(48 * sizeof(float));
    _Float16* Xf16   = (_Float16*)take((size_t)b * KPAD * sizeof(_Float16));
    _Float16* Bfrag  = (_Float16*)take(BFRAG_ELEMS * sizeof(_Float16));

    // 1) pose PCA + Rodrigues + WMMA A-matrix
    k_prep_pose<<<(b + 63) / 64, 64, 0, stream>>>(coeffs, betas, SC, hmean,
                                                  rotws, Xf16, b);
    // 2) joint regression precompute (once)
    k_jreduce<<<1, 544, 0, stream>>>(JR, shapedirs, vtempl, Jshape, Jtempl);
    // 3) pack B fragments (once)
    k_prepB<<<(int)((BFRAG_ELEMS + 255) / 256), 256, 0, stream>>>(shapedirs,
                                                                  posedirs, Bfrag);
    // 4) FK chain -> skinning matrices
    k_fk<<<(b + 63) / 64, 64, 0, stream>>>(betas, rotws, Jshape, Jtempl,
                                           Gws, jtrws, b);
    // 5) fused WMMA GEMM + skinning
    dim3 grid((b + 15) / 16, VTILES);
    k_gemm_skin<<<grid, 32, 0, stream>>>(Xf16, Bfrag, vtempl, weights, Gws,
                                         trans, (float*)d_out, b);
    // 6) jtr reorder + tips
    k_jtr<<<(b * 21 + 255) / 256, 256, 0, stream>>>(jtrws, trans,
                                                    (float*)d_out, b);
}
